// MultiViewPointFusion_91225105367648
// MI455X (gfx1250) — matrix operand order repaired and checked by reading, not compiled, not run on hardware
//
#include <hip/hip_runtime.h>
#include <math.h>

// ---------------------------------------------------------------------------
// MI455X (gfx1250) implementation.
// Precision: f16 operands + f32 accumulation via V_WMMA_F32_16X16X32_F16.
// Rationale: 36 GFLOP vs ~180MB traffic -> near BW-bound only if we use the
// K=32 f16 WMMA path (the f32 K=4 path is ~8x slower per instruction).
// 192MB L2 holds the whole working set, so weights/feature maps are streamed
// from L2 directly; LDS is used for the shared per-tile A panel in the fused
// gather+GEMM kernel.
// ---------------------------------------------------------------------------

typedef __attribute__((ext_vector_type(16))) _Float16 v16h;
typedef __attribute__((ext_vector_type(8)))  _Float16 v8h;
typedef __attribute__((ext_vector_type(8)))  float    v8f;

#define NPTS   200000
#define WPADF  800.0f
#define HPADF  448.0f
#define BNEPS  0.001f

static __device__ __forceinline__ v16h cat8(v8h lo, v8h hi) {
  return __builtin_shufflevector(lo, hi, 0,1,2,3,4,5,6,7,8,9,10,11,12,13,14,15);
}

static __device__ __forceinline__ v8f wmma_f16(v16h a, v16h b, v8f c) {
  // D = A(16x32) * B(32x16) + C ; f32 accumulate
  return __builtin_amdgcn_wmma_f32_16x16x32_f16(
      /*neg_a=*/false, a, /*neg_b=*/false, b,
      /*c_mod=*/(short)0, c, /*reuse_a=*/false, /*reuse_b=*/false);
}

// A fragment (16x32 f16): lane (l&15)=M row, g=l>>4 selects K runs
// [8g,8g+8) and [16+8g,16+8g+8) — per CDNA5 ISA 16-bit A layout.
static __device__ __forceinline__ v16h load_a_frag(const _Float16* row, int g) {
  return cat8(*(const v8h*)(row + 8*g), *(const v8h*)(row + 16 + 8*g));
}

// B fragment (32x16 f16): lane (l&15)=N col, g selects K runs
// [8g,8g+8) and [16+8g,16+8g+8) — same K order as the A fragment.
static __device__ __forceinline__ v16h load_b_frag(const _Float16* col, int g) {
  return cat8(*(const v8h*)(col + 8*g), *(const v8h*)(col + 16 + 8*g));
}

// ---------------------------------------------------------------------------
// Prep: NCHW f32 -> padded NHWC f16 (zero 1-px border) for one level.
// ---------------------------------------------------------------------------
__global__ void pad_nhwc_kernel(const float* __restrict__ feat,
                                _Float16* __restrict__ out, int H, int W) {
  size_t idx = (size_t)blockIdx.x * blockDim.x + threadIdx.x;
  size_t total = (size_t)6 * (H + 2) * (W + 2) * 128;
  if (idx >= total) return;
  int c = (int)(idx & 127);
  size_t r = idx >> 7;
  int x = (int)(r % (size_t)(W + 2)); r /= (size_t)(W + 2);
  int y = (int)(r % (size_t)(H + 2));
  int v = (int)(r / (size_t)(H + 2));
  int yi = y - 1, xi = x - 1;
  float val = 0.0f;
  if (yi >= 0 && yi < H && xi >= 0 && xi < W)
    val = feat[(((size_t)v * 128 + c) * H + yi) * W + xi];
  out[idx] = (_Float16)val;
}

// ---------------------------------------------------------------------------
// Prep: lat_w (L, O=128, C=128, 3, 3) f32 -> wf16 [lvl][tap][o][c] f16.
// ---------------------------------------------------------------------------
__global__ void wprep_kernel(const float* __restrict__ latw,
                             _Float16* __restrict__ wf) {
  int idx = blockIdx.x * 256 + threadIdx.x;
  if (idx >= 3 * 9 * 128 * 128) return;
  int c = idx & 127;
  int r = idx >> 7;
  int o = r & 127; r >>= 7;
  int tap = r % 9;
  int lvl = r / 9;
  int ky = tap / 3, kx = tap % 3;
  wf[idx] = (_Float16)latw[((((size_t)lvl * 128 + o) * 128 + c) * 3 + ky) * 3 + kx];
}

// ---------------------------------------------------------------------------
// Prep: fold BN into fused weight matrix W'[128][448] f16 + bias[128] f32.
// out = relu(bn1(img@W1^T) + bn2(pts@W2^T)) == relu([img|pts]@W'^T + bias)
// with W1 rows scaled by s1, W2 rows by s2 (exact algebra, done in f32).
// ---------------------------------------------------------------------------
__global__ void fusew_kernel(const float* __restrict__ w_img,
                             const float* __restrict__ bn_img,
                             const float* __restrict__ w_pts,
                             const float* __restrict__ bn_pts,
                             _Float16* __restrict__ Wp,
                             float* __restrict__ bias) {
  int j = threadIdx.x;  // 128 threads
  float s1 = bn_img[j] * rsqrtf(bn_img[384 + j] + BNEPS);
  float t1 = bn_img[128 + j] - bn_img[256 + j] * s1;
  float s2 = bn_pts[j] * rsqrtf(bn_pts[384 + j] + BNEPS);
  float t2 = bn_pts[128 + j] - bn_pts[256 + j] * s2;
  bias[j] = t1 + t2;
  for (int k = 0; k < 384; ++k)
    Wp[(size_t)j * 448 + k] = (_Float16)(w_img[(size_t)j * 384 + k] * s1);
  for (int k = 0; k < 64; ++k)
    Wp[(size_t)j * 448 + 384 + k] = (_Float16)(w_pts[(size_t)j * 64 + k] * s2);
}

// ---------------------------------------------------------------------------
// Point projection: pick first valid view, pixel coords, valid mask.
// ---------------------------------------------------------------------------
__global__ void project_kernel(const float* __restrict__ pts,
                               const float* __restrict__ l2i,
                               int* __restrict__ vidx,
                               float* __restrict__ xp,
                               float* __restrict__ yp,
                               float* __restrict__ valid, int n) {
  int i = blockIdx.x * 256 + threadIdx.x;
  if (i >= n) return;
  float X = pts[(size_t)i * 3 + 0];
  float Y = pts[(size_t)i * 3 + 1];
  float Z = pts[(size_t)i * 3 + 2];
  int best = 0;
  float bx = 0.0f, by = 0.0f;
  bool found = false;
  #pragma unroll
  for (int v = 0; v < 6; ++v) {
    const float* m = l2i + v * 16;
    float px = m[0] * X + m[1] * Y + m[2]  * Z + m[3];
    float py = m[4] * X + m[5] * Y + m[6]  * Z + m[7];
    float pz = m[8] * X + m[9] * Y + m[10] * Z + m[11];
    float x = px / pz, y = py / pz;
    bool ok = (x <= WPADF) && (y <= HPADF) && (x >= 0.0f) && (y >= 0.0f);
    if (ok && !found) { found = true; best = v; bx = x; by = y; }
  }
  vidx[i]  = best;
  xp[i]    = found ? bx : 0.0f;
  yp[i]    = found ? by : 0.0f;
  valid[i] = found ? 1.0f : 0.0f;
}

// ---------------------------------------------------------------------------
// Implicit-GEMM 3x3 conv via WMMA. One block = 16 pixels x 128 out-ch.
// Wave w computes out-ch tile [16w,16w+16). K = 9 taps * 128 ch, 36 WMMAs.
// A = weights [o][c] (L2-resident), B = padded NHWC f16 input.
// Output lateral features in NHWC f16 (8 contiguous channels per lane store).
// ---------------------------------------------------------------------------
__global__ void __launch_bounds__(256) conv_lat_kernel(
    const _Float16* __restrict__ inPad,   // (6, H+2, W+2, 128) f16
    const _Float16* __restrict__ wf,      // (9, 128, 128) f16 for this level
    const float*    __restrict__ cbias,   // (128,) f32
    _Float16*       __restrict__ out,     // (6, H, W, 128) f16
    int H, int W, int tilesPerView) {
  const int HW  = H * W;
  const int Wp2 = W + 2;
  int view = blockIdx.x / tilesPerView;
  int tile = blockIdx.x % tilesPerView;
  int p0   = tile * 16;
  int lane = threadIdx.x & 31;
  int wv   = threadIdx.x >> 5;
  int g    = lane >> 4;
  int nId  = lane & 15;

  int pix = p0 + nId;
  if (pix >= HW) pix = HW - 1;             // tail clamp (duplicate-safe)
  int py = pix / W, px = pix % W;

  const _Float16* inV = inPad + (size_t)view * (H + 2) * Wp2 * 128;

  v8f acc = {};
  #pragma unroll
  for (int tap = 0; tap < 9; ++tap) {
    int ky = tap / 3, kx = tap % 3;
    const _Float16* src = inV + ((size_t)(py + ky) * Wp2 + (px + kx)) * 128;
    const _Float16* wr  = wf + ((size_t)tap * 128 + 16 * wv + nId) * 128;
    #pragma unroll
    for (int c0 = 0; c0 < 128; c0 += 32) {
      v16h a = load_a_frag(wr + c0, g);
      v16h b = load_b_frag(src + c0, g);
      acc = wmma_f16(a, b, acc);
    }
  }

  const float* bb = cbias + 16 * wv + 8 * g;
  v8h h;
  #pragma unroll
  for (int r = 0; r < 8; ++r) h[r] = (_Float16)(acc[r] + bb[r]);
  *(v8h*)(out + ((size_t)view * HW + pix) * 128 + 16 * wv + 8 * g) = h;
}

// ---------------------------------------------------------------------------
// Bilinear sample of one level into 8 channels -> f16 LDS A panel.
// ---------------------------------------------------------------------------
static __device__ __forceinline__ void sample_level(
    const _Float16* __restrict__ lat, int Hl, int Wl,
    int vi, float xpv, float ypv, float va, int sub, _Float16* dst) {
  float fx = xpv * ((float)(Wl - 1) / WPADF);
  float fy = ypv * ((float)(Hl - 1) / HPADF);
  float x0 = floorf(fx), y0 = floorf(fy);
  int x0i = (int)x0, y0i = (int)y0;
  float wx1 = fx - x0, wy1 = fy - y0;
  float a[8];
  #pragma unroll
  for (int i = 0; i < 8; ++i) a[i] = 0.0f;
  #pragma unroll
  for (int dy = 0; dy < 2; ++dy) {
    float wy = dy ? wy1 : 1.0f - wy1;
    int yc = y0i + dy;
    bool oky = (yc >= 0) && (yc < Hl);
    int ycc = yc < 0 ? 0 : (yc >= Hl ? Hl - 1 : yc);
    #pragma unroll
    for (int dx = 0; dx < 2; ++dx) {
      float wx = dx ? wx1 : 1.0f - wx1;
      int xc = x0i + dx;
      bool okx = (xc >= 0) && (xc < Wl);
      int xcc = xc < 0 ? 0 : (xc >= Wl ? Wl - 1 : xc);
      float w = wy * wx * ((oky && okx) ? va : 0.0f);
      v8h v = *(const v8h*)(lat + (((size_t)vi * Hl + ycc) * Wl + xcc) * 128 + sub * 8);
      #pragma unroll
      for (int i = 0; i < 8; ++i) a[i] += w * (float)v[i];
    }
  }
  v8h h;
  #pragma unroll
  for (int i = 0; i < 8; ++i) h[i] = (_Float16)a[i];
  *(v8h*)dst = h;
}

// ---------------------------------------------------------------------------
// Fused gather + GEMM + BN + ReLU. Block = 16 points.
// Phase 1: 16 threads/point gather 3 levels + pts feats into LDS A (16x448).
// Phase 2: 8 waves; wave w does D(16pts x 16ch) over K=448 -> 14 WMMAs.
// ---------------------------------------------------------------------------
__global__ void __launch_bounds__(256) fuse_gemm_kernel(
    const _Float16* __restrict__ lat0, const _Float16* __restrict__ lat1,
    const _Float16* __restrict__ lat2,
    const int*   __restrict__ vidx, const float* __restrict__ xpv,
    const float* __restrict__ ypv,  const float* __restrict__ validv,
    const float* __restrict__ ptsf,
    const _Float16* __restrict__ Wp,   // (128, 448) f16 fused weights
    const float*    __restrict__ bias, // (128,) f32
    float* __restrict__ out) {
  __shared__ _Float16 As[16][480];     // K stride padded 448->480 (32B aligned frags)

  const int t  = threadIdx.x;
  const int p0 = blockIdx.x * 16;

  {  // ---- gather phase: 16 threads per point, 8 channels per thread/level
    int pl  = t >> 4;
    int sub = t & 15;
    int pt  = p0 + pl;
    int   vi = vidx[pt];
    float x  = xpv[pt];
    float y  = ypv[pt];
    float va = validv[pt];
    sample_level(lat0, 56, 100, vi, x, y, va, sub, &As[pl][0]);
    sample_level(lat1, 28,  50, vi, x, y, va, sub, &As[pl][128]);
    sample_level(lat2, 14,  25, vi, x, y, va, sub, &As[pl][256]);
    const float* pf = ptsf + (size_t)pt * 64 + sub * 4;
    _Float16* dst = &As[pl][384 + sub * 4];
    #pragma unroll
    for (int i = 0; i < 4; ++i) dst[i] = (_Float16)pf[i];
  }
  __syncthreads();

  // ---- GEMM phase
  int lane = t & 31, wv = t >> 5;
  int g = lane >> 4, nId = lane & 15;
  const _Float16* arow = &As[nId][0];
  const _Float16* brow = Wp + (size_t)(16 * wv + nId) * 448;
  __builtin_prefetch(brow, 0, 0);  // global_prefetch_b8 (L2-resident weights)

  v8f acc = {};
  #pragma unroll
  for (int kc = 0; kc < 448; kc += 32) {
    v16h a = load_a_frag(arow + kc, g);   // from LDS
    v16h b = load_b_frag(brow + kc, g);   // from global/L2
    acc = wmma_f16(a, b, acc);
  }

  float bs = bias[16 * wv + nId];
  #pragma unroll
  for (int r = 0; r < 8; ++r) {
    int pt = p0 + r + 8 * g;                        // C/D layout: M = r + 8g
    float v = acc[r] + bs;
    out[(size_t)pt * 128 + 16 * wv + nId] = v > 0.0f ? v : 0.0f;
  }
}

// ---------------------------------------------------------------------------
// Host launch
// ---------------------------------------------------------------------------
extern "C" void kernel_launch(void* const* d_in, const int* in_sizes, int n_in,
                              void* d_out, int out_size, void* d_ws, size_t ws_size,
                              hipStream_t stream) {
  (void)in_sizes; (void)n_in; (void)out_size; (void)ws_size;

  const float* img0  = (const float*)d_in[0];
  const float* img1  = (const float*)d_in[1];
  const float* img2  = (const float*)d_in[2];
  const float* pts   = (const float*)d_in[3];
  const float* ptsf  = (const float*)d_in[4];
  const float* l2i   = (const float*)d_in[5];
  const float* latw  = (const float*)d_in[6];
  const float* latb  = (const float*)d_in[7];
  const float* wimg  = (const float*)d_in[8];
  const float* bnimg = (const float*)d_in[9];
  const float* wpts  = (const float*)d_in[10];
  const float* bnpts = (const float*)d_in[11];
  float* out = (float*)d_out;

  char* ws = (char*)d_ws;
  size_t off = 0;
  auto take = [&](size_t bytes) -> void* {
    void* p = ws + off;
    off = (off + bytes + 255) & ~(size_t)255;
    return p;
  };

  _Float16* pad0 = (_Float16*)take((size_t)6 * 58 * 102 * 128 * 2);
  _Float16* pad1 = (_Float16*)take((size_t)6 * 30 *  52 * 128 * 2);
  _Float16* pad2 = (_Float16*)take((size_t)6 * 16 *  27 * 128 * 2);
  _Float16* lat0 = (_Float16*)take((size_t)6 * 5600 * 128 * 2);
  _Float16* lat1 = (_Float16*)take((size_t)6 * 1400 * 128 * 2);
  _Float16* lat2 = (_Float16*)take((size_t)6 *  350 * 128 * 2);
  _Float16* wf   = (_Float16*)take((size_t)3 * 9 * 128 * 128 * 2);
  _Float16* Wp   = (_Float16*)take((size_t)128 * 448 * 2);
  float*    bias = (float*)take(128 * 4);
  int*      vidx = (int*)take((size_t)NPTS * 4);
  float*    xp   = (float*)take((size_t)NPTS * 4);
  float*    yp   = (float*)take((size_t)NPTS * 4);
  float*    vld  = (float*)take((size_t)NPTS * 4);

  dim3 blk(256);

  size_t t0 = (size_t)6 * 58 * 102 * 128;
  size_t t1 = (size_t)6 * 30 *  52 * 128;
  size_t t2 = (size_t)6 * 16 *  27 * 128;
  pad_nhwc_kernel<<<(unsigned)((t0 + 255) / 256), blk, 0, stream>>>(img0, pad0, 56, 100);
  pad_nhwc_kernel<<<(unsigned)((t1 + 255) / 256), blk, 0, stream>>>(img1, pad1, 28, 50);
  pad_nhwc_kernel<<<(unsigned)((t2 + 255) / 256), blk, 0, stream>>>(img2, pad2, 14, 25);

  wprep_kernel<<<(3 * 9 * 128 * 128 + 255) / 256, blk, 0, stream>>>(latw, wf);
  fusew_kernel<<<1, 128, 0, stream>>>(wimg, bnimg, wpts, bnpts, Wp, bias);
  project_kernel<<<(NPTS + 255) / 256, blk, 0, stream>>>(pts, l2i, vidx, xp, yp, vld, NPTS);

  conv_lat_kernel<<<6 * 350, blk, 0, stream>>>(pad0, wf,                 latb,       lat0, 56, 100, 350);
  conv_lat_kernel<<<6 *  88, blk, 0, stream>>>(pad1, wf + 9 * 128 * 128, latb + 128, lat1, 28,  50,  88);
  conv_lat_kernel<<<6 *  22, blk, 0, stream>>>(pad2, wf + 2 * 9 * 128 * 128, latb + 256, lat2, 14, 25, 22);

  fuse_gemm_kernel<<<NPTS / 16, blk, 0, stream>>>(
      lat0, lat1, lat2, vidx, xp, yp, vld, ptsf, Wp, bias, out);
}
